// GlobalPointer_64673617543665
// MI455X (gfx1250) — matrix operand order repaired and checked by reading, not compiled, not run on hardware
//
#include <hip/hip_runtime.h>
#include <hip/hip_bf16.h>

// ---------------------------------------------------------------------------
// Problem constants (from reference)
// ---------------------------------------------------------------------------
#define BATCH   32
#define SEQ     512
#define HDIM    1024
#define HEADS   12
#define HSZ     64
#define NPROJ   (HEADS * 2 * HSZ)   // 1536
#define MROWS   (BATCH * SEQ)       // 16384
#define INF_F   1.0e13f

typedef __bf16 bf16_t;
typedef bf16_t v16bf __attribute__((ext_vector_type(16)));
typedef float  v8f   __attribute__((ext_vector_type(8)));
typedef int    vec4i __attribute__((vector_size(4 * sizeof(int))));

union FragBF { unsigned int u[8]; v16bf v; };

// CDNA5 async global->LDS copy (ASYNCcnt path), guarded so absence of the
// builtin on a given toolchain falls back to the register-staged copy.
#if defined(__has_builtin)
#if __has_builtin(__builtin_amdgcn_global_load_async_to_lds_b128)
#define HAVE_ASYNC_LDS 1
#endif
#if __has_builtin(__builtin_amdgcn_s_wait_asynccnt)
#define HAVE_WAIT_ASYNC 1
#endif
#endif

__device__ __forceinline__ void copy16_g2lds(const unsigned short* __restrict__ g,
                                             unsigned short* l) {
#ifdef HAVE_ASYNC_LDS
    __builtin_amdgcn_global_load_async_to_lds_b128((vec4i*)g, (vec4i*)l, 0, 0);
#else
    *(uint4*)l = *(const uint4*)g;
#endif
}
__device__ __forceinline__ void wait_async_all() {
#ifdef HAVE_WAIT_ASYNC
    __builtin_amdgcn_s_wait_asynccnt(0);
#endif
}

// float -> bf16 (round-to-nearest-even), bf16(ushort) -> float
__device__ __forceinline__ unsigned short f2bf(float f) {
    unsigned int u = __float_as_uint(f);
    unsigned int r = u + 0x7FFFu + ((u >> 16) & 1u);
    return (unsigned short)(r >> 16);
}
__device__ __forceinline__ float bf2f(unsigned short s) {
    return __uint_as_float(((unsigned int)s) << 16);
}

// A-fragment (16x32 bf16): lane half selects K-group; VGPR v holds a K-pair.
// dword index within a K-contiguous row of packed bf16 pairs:
__device__ __forceinline__ int dwA(int v, int half) {
    return v + 8 * half;       // K-pair = 2*dword
}
// B-fragment (32x16 bf16): lanes 0-15 hold K=0..15 pairs, lanes 16-31 K=16..31.
__device__ __forceinline__ int dwB(int v, int half) {
    return v + 8 * half;
}

// ---------------------------------------------------------------------------
// Kernel 0: fp32 -> bf16 conversion, 4 elements per thread
// ---------------------------------------------------------------------------
__global__ void cvt_f32_to_bf16_x4(const float* __restrict__ in,
                                   unsigned short* __restrict__ out, int n4) {
    int i = blockIdx.x * blockDim.x + threadIdx.x;
    if (i >= n4) return;
    float4 f = ((const float4*)in)[i];
    unsigned int lo = (unsigned int)f2bf(f.x) | ((unsigned int)f2bf(f.y) << 16);
    unsigned int hi = (unsigned int)f2bf(f.z) | ((unsigned int)f2bf(f.w) << 16);
    ((uint2*)out)[i] = make_uint2(lo, hi);
}

// ---------------------------------------------------------------------------
// Kernel 1: projection GEMM  proj = x @ W + b   (bf16 in, bf16 out)
//   M=16384, K=1024, N=1536.  Block tile 128x128, 8 waves (2x4), wave 64x32.
//   x tile staged with async global->LDS b128; W tile staged transposed
//   (LDS [n][k]) so WMMA B-fragments are packed bf16 K-pair dword reads.
// ---------------------------------------------------------------------------
#define G1_LDS_STRIDE 40   // 32 K elems + 8 pad (bf16 units); 20 dwords

__global__ __launch_bounds__(256) void gemm_proj_bias(
        const unsigned short* __restrict__ xb,
        const unsigned short* __restrict__ wb,
        const float* __restrict__ bias,
        unsigned short* __restrict__ projb) {
    __shared__ __align__(16) unsigned short sX[128 * G1_LDS_STRIDE];
    __shared__ __align__(16) unsigned short sW[128 * G1_LDS_STRIDE];

    const int tid  = threadIdx.x;
    const int m0   = blockIdx.y * 128;
    const int n0   = blockIdx.x * 128;
    const int lane = tid & 31;
    const int wid  = tid >> 5;
    const int wm   = wid >> 2;        // 0..1 -> 64-row slab
    const int wn   = wid & 3;         // 0..3 -> 32-col slab
    const int half = lane >> 4;
    const int l16  = lane & 15;

    v8f acc[4][2];
    #pragma unroll
    for (int tm = 0; tm < 4; ++tm)
        #pragma unroll
        for (int tn = 0; tn < 2; ++tn) acc[tm][tn] = {};

    const unsigned int* x32 = (const unsigned int*)sX;
    const unsigned int* w32 = (const unsigned int*)sW;

    for (int k0 = 0; k0 < HDIM; k0 += 32) {
        // ---- stage x tile: 128 rows x 32 K, async b128 copies ----
        #pragma unroll
        for (int j = 0; j < 2; ++j) {
            int c   = tid * 2 + j;            // 0..511
            int row = c >> 2;
            int col = (c & 3) * 8;
            copy16_g2lds(xb + (size_t)(m0 + row) * HDIM + k0 + col,
                         sX + row * G1_LDS_STRIDE + col);
        }
        // ---- stage W tile transposed: LDS[n][k], 32 K-rows x 128 N ----
        #pragma unroll
        for (int j = 0; j < 2; ++j) {
            int c  = tid * 2 + j;             // 0..511
            int kr = c >> 4;                  // 0..31
            int nc = (c & 15) * 8;            // 0..120
            uint4 v = *(const uint4*)(wb + (size_t)(k0 + kr) * NPROJ + n0 + nc);
            const unsigned short* e = (const unsigned short*)&v;
            #pragma unroll
            for (int i = 0; i < 8; ++i)
                sW[(nc + i) * G1_LDS_STRIDE + kr] = e[i];
        }
        // prefetch next K slab of x into L2 (global_prefetch_b8)
        if (k0 + 32 < HDIM) {
            __builtin_prefetch(xb + (size_t)(m0 + (tid >> 1)) * HDIM + k0 + 32, 0, 1);
        }
        wait_async_all();
        __syncthreads();

        // ---- fragments + 8 WMMA per wave ----
        FragBF a[4], b[2];
        #pragma unroll
        for (int tm = 0; tm < 4; ++tm) {
            int r = wm * 64 + tm * 16 + l16;
            #pragma unroll
            for (int v = 0; v < 8; ++v)
                a[tm].u[v] = x32[r * (G1_LDS_STRIDE / 2) + dwA(v, half)];
        }
        #pragma unroll
        for (int tn = 0; tn < 2; ++tn) {
            int n = wn * 32 + tn * 16 + l16;
            #pragma unroll
            for (int v = 0; v < 8; ++v)
                b[tn].u[v] = w32[n * (G1_LDS_STRIDE / 2) + dwB(v, half)];
        }
        #pragma unroll
        for (int tm = 0; tm < 4; ++tm)
            #pragma unroll
            for (int tn = 0; tn < 2; ++tn)
                acc[tm][tn] = __builtin_amdgcn_wmma_f32_16x16x32_bf16(
                    false, a[tm].v, false, b[tn].v,
                    (short)0, acc[tm][tn], false, false);
        __syncthreads();
    }

    // ---- epilogue: + bias, convert to bf16, store proj ----
    #pragma unroll
    for (int tn = 0; tn < 2; ++tn) {
        int n  = n0 + wn * 32 + tn * 16 + l16;   // C/D layout: col = lane%16
        float bn = bias[n];
        #pragma unroll
        for (int tm = 0; tm < 4; ++tm) {
            #pragma unroll
            for (int v = 0; v < 8; ++v) {
                int m = m0 + wm * 64 + tm * 16 + v + 8 * half; // row = v + 8*(lane/16)
                projb[(size_t)m * NPROJ + n] = f2bf(acc[tm][tn][v] + bn);
            }
        }
    }
}

// ---------------------------------------------------------------------------
// Kernel 2: RoPE + split into q[b,h,s,64], k[b,h,s,64] (bf16)
// ---------------------------------------------------------------------------
__global__ void rope_split(const unsigned short* __restrict__ projb,
                           unsigned short* __restrict__ qb,
                           unsigned short* __restrict__ kb) {
    int id = blockIdx.x * blockDim.x + threadIdx.x;
    const int total = BATCH * SEQ * HEADS * (HSZ / 2);
    if (id >= total) return;
    int i    = id & 31;            // pair index 0..31
    int rest = id >> 5;
    int h    = rest % HEADS;  rest /= HEADS;
    int s    = rest % SEQ;
    int b    = rest / SEQ;

    // inv_freq = 10000^(-2i/64) ; ln(10000) = 9.210340371976184
    float inv = __expf(-((float)(2 * i) * (1.0f / 64.0f)) * 9.210340371976184f);
    float ang = (float)s * inv;
    float sn, cs;
    __sincosf(ang, &sn, &cs);

    size_t pbase = ((size_t)(b * SEQ + s)) * NPROJ + h * (2 * HSZ);
    float q1 = bf2f(projb[pbase + 2 * i]);
    float q2 = bf2f(projb[pbase + 2 * i + 1]);
    float k1 = bf2f(projb[pbase + HSZ + 2 * i]);
    float k2 = bf2f(projb[pbase + HSZ + 2 * i + 1]);

    size_t obase = (((size_t)(b * HEADS + h) * SEQ + s) * HSZ);
    qb[obase + 2 * i]     = f2bf(q1 * cs - q2 * sn);
    qb[obase + 2 * i + 1] = f2bf(q1 * sn + q2 * cs);
    kb[obase + 2 * i]     = f2bf(k1 * cs - k2 * sn);
    kb[obase + 2 * i + 1] = f2bf(k1 * sn + k2 * cs);
}

// ---------------------------------------------------------------------------
// Kernel 3: attention logits per (b,h): q(512x64) @ k^T -> 512x512, fused
//   pad-mask, causal(tril m>n) mask and 1/sqrt(64) scale. Block tile 128x128.
//   Tiles staged with async global->LDS b128 copies.
// ---------------------------------------------------------------------------
#define G2_LDS_STRIDE 72   // 64 + 8 pad (bf16 units); 36 dwords

__global__ __launch_bounds__(256) void attn_logits(
        const unsigned short* __restrict__ qb,
        const unsigned short* __restrict__ kb,
        const int* __restrict__ am,
        float* __restrict__ out) {
    __shared__ __align__(16) unsigned short sQ[128 * G2_LDS_STRIDE];
    __shared__ __align__(16) unsigned short sK[128 * G2_LDS_STRIDE];

    const int bh   = blockIdx.z;           // b*HEADS + h
    const int b    = bh / HEADS;
    const int m0   = blockIdx.y * 128;
    const int n0   = blockIdx.x * 128;
    const int tid  = threadIdx.x;
    const int lane = tid & 31;
    const int wid  = tid >> 5;
    const int wm   = wid >> 2;
    const int wn   = wid & 3;
    const int half = lane >> 4;
    const int l16  = lane & 15;

    // ---- stage q / k tiles: 128 rows x 64 d each, async b128 copies ----
    #pragma unroll
    for (int j = 0; j < 4; ++j) {
        int c   = tid + j * 256;          // 0..1023
        int row = c >> 3;
        int col = (c & 7) * 8;
        copy16_g2lds(qb + ((size_t)bh * SEQ + m0 + row) * HSZ + col,
                     sQ + row * G2_LDS_STRIDE + col);
        copy16_g2lds(kb + ((size_t)bh * SEQ + n0 + row) * HSZ + col,
                     sK + row * G2_LDS_STRIDE + col);
    }
    wait_async_all();
    __syncthreads();

    const unsigned int* q32 = (const unsigned int*)sQ;
    const unsigned int* k32 = (const unsigned int*)sK;

    v8f acc[4][2];
    #pragma unroll
    for (int tm = 0; tm < 4; ++tm)
        #pragma unroll
        for (int tn = 0; tn < 2; ++tn) acc[tm][tn] = {};

    #pragma unroll
    for (int kc = 0; kc < 2; ++kc) {      // K = 64 -> two 32-wide steps
        const int kOff = kc * 16;         // dwords
        FragBF a[4], bfr[2];
        #pragma unroll
        for (int tm = 0; tm < 4; ++tm) {
            int r = wm * 64 + tm * 16 + l16;
            #pragma unroll
            for (int v = 0; v < 8; ++v)
                a[tm].u[v] = q32[r * (G2_LDS_STRIDE / 2) + kOff + dwA(v, half)];
        }
        #pragma unroll
        for (int tn = 0; tn < 2; ++tn) {
            int n = wn * 32 + tn * 16 + l16;
            #pragma unroll
            for (int v = 0; v < 8; ++v)
                bfr[tn].u[v] = k32[n * (G2_LDS_STRIDE / 2) + kOff + dwB(v, half)];
        }
        #pragma unroll
        for (int tm = 0; tm < 4; ++tm)
            #pragma unroll
            for (int tn = 0; tn < 2; ++tn)
                acc[tm][tn] = __builtin_amdgcn_wmma_f32_16x16x32_bf16(
                    false, a[tm].v, false, bfr[tn].v,
                    (short)0, acc[tm][tn], false, false);
    }

    // ---- epilogue: masks + scale, single fp32 write of logits ----
    int amn[2];
    #pragma unroll
    for (int tn = 0; tn < 2; ++tn)
        amn[tn] = am[b * SEQ + n0 + wn * 32 + tn * 16 + l16];

    #pragma unroll
    for (int tm = 0; tm < 4; ++tm) {
        #pragma unroll
        for (int v = 0; v < 8; ++v) {
            int m   = m0 + wm * 64 + tm * 16 + v + 8 * half;
            int amm = am[b * SEQ + m];
            #pragma unroll
            for (int tn = 0; tn < 2; ++tn) {
                int n = n0 + wn * 32 + tn * 16 + l16;
                float maskf = (float)(1 - amm * amn[tn]) + ((m > n) ? 1.0f : 0.0f);
                float val   = (acc[tm][tn][v] - maskf * INF_F) * 0.125f;
                out[((size_t)bh * SEQ + m) * SEQ + n] = val;
            }
        }
    }
}

// ---------------------------------------------------------------------------
// launch
// ---------------------------------------------------------------------------
extern "C" void kernel_launch(void* const* d_in, const int* in_sizes, int n_in,
                              void* d_out, int out_size, void* d_ws, size_t ws_size,
                              hipStream_t stream) {
    const float* x    = (const float*)d_in[0];   // (32,512,1024)
    const float* W    = (const float*)d_in[1];   // (1024,1536)
    const float* bias = (const float*)d_in[2];   // (1536,)
    const int*   am   = (const int*)d_in[3];     // (32,512)
    float* out        = (float*)d_out;           // (32,12,512,512)

    // workspace layout (bf16 buffers)
    unsigned short* xb    = (unsigned short*)d_ws;                       // 16384*1024
    unsigned short* wb    = xb    + (size_t)MROWS * HDIM;                // 1024*1536
    unsigned short* projb = wb    + (size_t)HDIM * NPROJ;                // 16384*1536
    unsigned short* qb    = projb + (size_t)MROWS * NPROJ;               // 32*12*512*64
    unsigned short* kb    = qb    + (size_t)BATCH * HEADS * SEQ * HSZ;

    // 0) convert inputs to bf16
    {
        int n4 = (MROWS * HDIM) / 4;
        cvt_f32_to_bf16_x4<<<(n4 + 255) / 256, 256, 0, stream>>>(x, xb, n4);
    }
    {
        int n4 = (HDIM * NPROJ) / 4;
        cvt_f32_to_bf16_x4<<<(n4 + 255) / 256, 256, 0, stream>>>(W, wb, n4);
    }

    // 1) projection GEMM + bias
    gemm_proj_bias<<<dim3(NPROJ / 128, MROWS / 128), 256, 0, stream>>>(
        xb, wb, bias, projb);

    // 2) RoPE + head split
    {
        int total = BATCH * SEQ * HEADS * (HSZ / 2);
        rope_split<<<(total + 255) / 256, 256, 0, stream>>>(projb, qb, kb);
    }

    // 3) attention logits with fused masks + scale
    attn_logits<<<dim3(SEQ / 128, SEQ / 128, BATCH * HEADS), 256, 0, stream>>>(
        qb, kb, am, out);
}
